// ShiftedSparseAttention_68994354643328
// MI455X (gfx1250) — matrix operand order repaired and checked by reading, not compiled, not run on hardware
//
#include <hip/hip_runtime.h>

// ---------------------------------------------------------------------------
// Shifted sparse (group) attention for MI455X / gfx1250.
// [1, 16, 8192, 128] fp32 -> roll(-512) -> per-1024-group softmax(QK^T/sqrt d)V
// Flash-attention over 64-key tiles, f16 WMMA w/ f32 accumulation,
// double-buffered LDS with software-pipelined global fetch (1 barrier/tile).
// ---------------------------------------------------------------------------

typedef _Float16 h4  __attribute__((ext_vector_type(4)));
typedef _Float16 h8  __attribute__((ext_vector_type(8)));
typedef _Float16 h16 __attribute__((ext_vector_type(16)));
typedef float    f4v __attribute__((ext_vector_type(4)));
typedef float    f8v __attribute__((ext_vector_type(8)));

#define HEADS  16
#define SEQ    8192
#define SMASK  8191
#define HD     128
#define GRP    1024
#define SHIFTN 512
#define TK     64              // keys per tile
#define NKT    (GRP / TK)      // 16 key tiles per group
#define QTILE  128             // query rows per block (8 waves x 16 rows)
#define SCALE  0.08838834764831845f   // 128^-0.5

// padded LDS row lengths (halfs); all fragment strides stay 16B-aligned
#define KPAD 136
#define VPAD 72
#define PPAD 72

// Build a 16-half fragment from two 16-byte LDS reads (ds_load_b128 x2).
__device__ __forceinline__ h16 lds_frag(const _Float16* p0, const _Float16* p1) {
  h8 lo = *(const h8*)p0;
  h8 hi = *(const h8*)p1;
  h16 r;
#pragma unroll
  for (int i = 0; i < 8; ++i) { r[i] = lo[i]; r[i + 8] = hi[i]; }
  return r;
}

// Load 8 consecutive f32 from global (2x b128), scale, convert to f16.
__device__ __forceinline__ h8 gld_scaled_h8(const float* p, float s) {
  f4v a = *(const f4v*)p;
  f4v b = *(const f4v*)(p + 4);
  h8 r;
  r[0] = (_Float16)(a[0] * s); r[1] = (_Float16)(a[1] * s);
  r[2] = (_Float16)(a[2] * s); r[3] = (_Float16)(a[3] * s);
  r[4] = (_Float16)(b[0] * s); r[5] = (_Float16)(b[1] * s);
  r[6] = (_Float16)(b[2] * s); r[7] = (_Float16)(b[3] * s);
  return r;
}

__global__ __launch_bounds__(256, 1)
void ssa_fwd(const float* __restrict__ Q, const float* __restrict__ K,
             const float* __restrict__ V, float* __restrict__ O) {
  // double-buffered: K tile [key][d] f16 ; V tile transposed [d][key] f16
  __shared__ __align__(16) _Float16 sK [2][TK][KPAD];
  __shared__ __align__(16) _Float16 sVT[2][HD][VPAD];
  // per-wave P scratch (C-layout -> A-layout crossing) [wave][row16][col64]
  __shared__ __align__(16) _Float16 sP [8][16][PPAD];

  const int gid  = blockIdx.x;           // 16 heads * 8 groups * 8 qtiles
  const int head = gid >> 6;
  const int grp  = (gid >> 3) & 7;
  const int qt   = gid & 7;

  const int t     = threadIdx.x;
  const int wave  = t >> 5;
  const int lane  = t & 31;
  const int lhalf = lane >> 4;           // 0: lanes 0-15, 1: lanes 16-31
  const int l16   = lane & 15;

  const size_t headOff = (size_t)head * SEQ * HD;
  const float* kbase = K + headOff;
  const float* vbase = V + headOff;

  // cooperative-load mapping: 4 threads per key row, 32 d each
  const int ldrow = t >> 2;
  const int ldd   = (t & 3) * 32;

  // ---- software-pipeline staging registers (f32, 64 VGPRs) ----------------
  f4v kreg[8], vreg[8];

  // issue global loads for key tile kt into staging registers (no waits here)
  auto issue_loads = [&](int kt) {
    const int krow = (grp * GRP + kt * TK + ldrow + SHIFTN) & SMASK;
    const float* kp = kbase + (size_t)krow * HD + ldd;
    const float* vp = vbase + (size_t)krow * HD + ldd;
#pragma unroll
    for (int j = 0; j < 8; ++j) {
      kreg[j] = *(const f4v*)(kp + j * 4);
      vreg[j] = *(const f4v*)(vp + j * 4);
    }
  };

  // convert staged f32 -> f16 and store into LDS buffer `buf`
  auto store_tile = [&](int buf) {
#pragma unroll
    for (int j = 0; j < 8; ++j) {
      const int d = ldd + j * 4;
      h4 kh;
      kh[0] = (_Float16)kreg[j][0]; kh[1] = (_Float16)kreg[j][1];
      kh[2] = (_Float16)kreg[j][2]; kh[3] = (_Float16)kreg[j][3];
      *(h4*)&sK[buf][ldrow][d] = kh;     // ds_store_b64
      sVT[buf][d + 0][ldrow] = (_Float16)vreg[j][0];
      sVT[buf][d + 1][ldrow] = (_Float16)vreg[j][1];
      sVT[buf][d + 2][ldrow] = (_Float16)vreg[j][2];
      sVT[buf][d + 3][ldrow] = (_Float16)vreg[j][3];
    }
  };

  // ---- Q A-fragments (16x32 f16 per 32-wide d chunk), pre-scaled ----------
  // A layout (ISA 7.12.2): lane = row(l16); halfs 0..7 -> K = dc*32+lhalf*8+i,
  // halfs 8..15 -> +16.
  const int qrow = (grp * GRP + qt * QTILE + wave * 16 + l16 + SHIFTN) & SMASK;
  const float* qp = Q + headOff + (size_t)qrow * HD;
  h16 qa[4];
#pragma unroll
  for (int dc = 0; dc < 4; ++dc) {
    const int d0 = dc * 32 + lhalf * 8;
    h8 lo = gld_scaled_h8(qp + d0, SCALE);
    h8 hi = gld_scaled_h8(qp + d0 + 16, SCALE);
#pragma unroll
    for (int i = 0; i < 8; ++i) { qa[dc][i] = lo[i]; qa[dc][i + 8] = hi[i]; }
  }

  // ---- flash-attention state (C-layout: lane owns rows lhalf*8 + r) -------
  float m[8], l[8];
  f8v   o[8];                            // 8 d-subtiles of 16 cols
#pragma unroll
  for (int r = 0; r < 8; ++r) { m[r] = -3.0e38f; l[r] = 0.0f; }
#pragma unroll
  for (int dt = 0; dt < 8; ++dt)
#pragma unroll
    for (int e = 0; e < 8; ++e) o[dt][e] = 0.0f;

  // ---- prologue: tile 0 into buffer 0 -------------------------------------
  issue_loads(0);
  store_tile(0);
  __syncthreads();

  for (int kt = 0; kt < NKT; ++kt) {
    const int cur = kt & 1;

    // kick off next tile's global fetch; it flies during the compute below
    if (kt + 1 < NKT) {
      if (kt + 2 < NKT) {                // warm L2 at distance 2
        const int prow = (grp * GRP + (kt + 2) * TK + ldrow + SHIFTN) & SMASK;
        __builtin_prefetch(kbase + (size_t)prow * HD + ldd, 0, 1);
        __builtin_prefetch(vbase + (size_t)prow * HD + ldd, 0, 1);
      }
      issue_loads(kt + 1);
    }

    // ---- S = Q K^T for 16 rows x 64 keys (4 x v_wmma per 16-key subtile) --
    f8v s[4];
#pragma unroll
    for (int nt = 0; nt < 4; ++nt) {
      f8v acc = {0.f, 0.f, 0.f, 0.f, 0.f, 0.f, 0.f, 0.f};
#pragma unroll
      for (int dc = 0; dc < 4; ++dc) {
        // B layout 32x16: lane = col(key = nt*16+l16); k = lhalf*8+i, +16 for halfs 8..15
        const _Float16* bp = &sK[cur][nt * 16 + l16][dc * 32 + lhalf * 8];
        h16 bk = lds_frag(bp, bp + 16);
        acc = __builtin_amdgcn_wmma_f32_16x16x32_f16(
                  false, qa[dc], false, bk, (short)0, acc, false, false);
      }
      s[nt] = acc;
    }

    // ---- online softmax (rows live per-lane; reduce across 16-lane group) -
    float al[8];
#pragma unroll
    for (int r = 0; r < 8; ++r) {
      float v0 = fmaxf(fmaxf(s[0][r], s[1][r]), fmaxf(s[2][r], s[3][r]));
      v0 = fmaxf(v0, __shfl_xor(v0, 1, 32));
      v0 = fmaxf(v0, __shfl_xor(v0, 2, 32));
      v0 = fmaxf(v0, __shfl_xor(v0, 4, 32));
      v0 = fmaxf(v0, __shfl_xor(v0, 8, 32));
      const float mn = fmaxf(m[r], v0);
      al[r] = __expf(m[r] - mn);
      m[r]  = mn;
    }
    float rs[8];
#pragma unroll
    for (int r = 0; r < 8; ++r) rs[r] = 0.0f;
#pragma unroll
    for (int nt = 0; nt < 4; ++nt)
#pragma unroll
      for (int r = 0; r < 8; ++r) {
        const float p = __expf(s[nt][r] - m[r]);
        s[nt][r] = p;
        rs[r] += p;
      }
#pragma unroll
    for (int r = 0; r < 8; ++r) {
      float v0 = rs[r];
      v0 += __shfl_xor(v0, 1, 32);
      v0 += __shfl_xor(v0, 2, 32);
      v0 += __shfl_xor(v0, 4, 32);
      v0 += __shfl_xor(v0, 8, 32);
      l[r] = l[r] * al[r] + v0;
#pragma unroll
      for (int dt = 0; dt < 8; ++dt) o[dt][r] *= al[r];
    }

    // ---- P: C-layout -> row-major f16 scratch (per-wave; no barrier) ------
#pragma unroll
    for (int nt = 0; nt < 4; ++nt)
#pragma unroll
      for (int r = 0; r < 8; ++r)
        sP[wave][lhalf * 8 + r][nt * 16 + l16] = (_Float16)s[nt][r];

    // ---- O += P V  (A = P 16x32 chunks, B = V^T subtiles) -----------------
#pragma unroll
    for (int kc = 0; kc < 2; ++kc) {
      const _Float16* pp = &sP[wave][l16][kc * 32 + lhalf * 8];
      h16 pa = lds_frag(pp, pp + 16);    // halves 8..15 at K+16
#pragma unroll
      for (int dt = 0; dt < 8; ++dt) {
        const _Float16* vb = &sVT[cur][dt * 16 + l16][kc * 32 + lhalf * 8];
        h16 bv = lds_frag(vb, vb + 16);
        o[dt] = __builtin_amdgcn_wmma_f32_16x16x32_f16(
                    false, pa, false, bv, (short)0, o[dt], false, false);
      }
    }

    // ---- drain staged loads into the other buffer; one barrier per tile ---
    if (kt + 1 < NKT) {
      store_tile(cur ^ 1);               // s_wait_loadcnt lands here, not above
      __syncthreads();
    }
  }

  // ---- finalize: divide by row sums, store with un-shift ------------------
  const int obase = grp * GRP + qt * QTILE + wave * 16;
#pragma unroll
  for (int r = 0; r < 8; ++r) {
    const float inv = 1.0f / l[r];
    const int row  = lhalf * 8 + r;
    const int orow = (obase + row + SHIFTN) & SMASK;
    float* op = O + headOff + (size_t)orow * HD + l16;
#pragma unroll
    for (int dt = 0; dt < 8; ++dt)
      op[dt * 16] = o[dt][r] * inv;
  }
}

extern "C" void kernel_launch(void* const* d_in, const int* in_sizes, int n_in,
                              void* d_out, int out_size, void* d_ws, size_t ws_size,
                              hipStream_t stream) {
  (void)in_sizes; (void)n_in; (void)out_size; (void)d_ws; (void)ws_size;
  const float* q = (const float*)d_in[0];
  const float* k = (const float*)d_in[1];
  const float* v = (const float*)d_in[2];
  float* out = (float*)d_out;
  const int blocks = HEADS * (SEQ / GRP) * (GRP / QTILE);   // 1024
  ssa_fwd<<<dim3(blocks), dim3(256), 0, stream>>>(q, k, v, out);
}
